// LearnedAttentionPattern_76982993814140
// MI455X (gfx1250) — matrix operand order repaired and checked by reading, not compiled, not run on hardware
//
#include <hip/hip_runtime.h>
#include <math.h>

#define NB 4
#define NS 2048
#define ND 1024
#define NH 4
#define NP 32
#define NM 8
#define QT (NS / 16)      // 128 16-row tiles along S
#define NOUT 140          // 128 feats + 8 sel + 4 comb
#define OTILES 9          // ceil(140/16)

typedef _Float16 half_t;
typedef __attribute__((ext_vector_type(16))) _Float16 v16h;
typedef __attribute__((ext_vector_type(8)))  _Float16 v8h;
typedef __attribute__((ext_vector_type(8)))  float    v8f;
typedef __attribute__((ext_vector_type(4)))  float    v4f;
typedef __attribute__((ext_vector_type(2)))  float    v2f;

union Frag16 { v16h v; v8h h[2]; };
union Out8   { float f[8]; v4f v[2]; };

// A-operand (16-bit 16x32): lane group g holds K in {g*8..g*8+7} U {16+g*8..16+g*8+7},
// row M = lane&15. Row-major [row][k] storage -> two contiguous 16B chunks.
__device__ __forceinline__ v16h load_afrag(const half_t* __restrict__ row, int g) {
    Frag16 f;
    f.h[0] = *(const v8h*)(row + g * 8);
    f.h[1] = *(const v8h*)(row + 16 + g * 8);
    return f.v;
}

// B-operand (16-bit 32x16): lane group g holds K in {g*8..g*8+7} U {16+g*8..16+g*8+7}, col N = lane&15.
// Row-major [col][k] storage -> two contiguous 16B chunks.
__device__ __forceinline__ v16h load_bfrag(const half_t* __restrict__ row, int g) {
    Frag16 f;
    f.h[0] = *(const v8h*)(row + g * 8);
    f.h[1] = *(const v8h*)(row + 16 + g * 8);
    return f.v;
}

// ---------------------------------------------------------------------------
// Kernel 1: fused feature GEMM Y[8192 x 140] = X[8192 x 1024] . W^T via
// v_wmma_f32_16x16x4_f32 (f32 precision). One wave per 16-token x 16-output tile.
// Outputs: feats (f16), raw selection logits, comb weights.
// ---------------------------------------------------------------------------
__global__ void k_feat_wmma(const float* __restrict__ x,
                            const float* __restrict__ gen_w, const float* __restrict__ gen_b,
                            const float* __restrict__ sel_w, const float* __restrict__ sel_b,
                            const float* __restrict__ comb_w, const float* __restrict__ comb_b,
                            half_t* __restrict__ feats, float* __restrict__ logits,
                            float* __restrict__ comb) {
    const int blk  = blockIdx.x;
    const int ot   = blk % OTILES;          // output tile
    const int tok0 = (blk / OTILES) * 16;   // token tile base
    const int lane = threadIdx.x & 31;
    const int g = lane >> 4, n = lane & 15;
    const int o = ot * 16 + n;              // this lane's output column

    const float* wrow;
    float bias;
    if (o < NH * NP)            { wrow = gen_w  + (size_t)o * ND;              bias = gen_b[o]; }
    else if (o < NH * NP + NM)  { wrow = sel_w  + (size_t)(o - 128) * ND;      bias = sel_b[o - 128]; }
    else if (o < NOUT)          { wrow = comb_w + (size_t)(o - 136) * ND;      bias = comb_b[o - 136]; }
    else                        { wrow = gen_w;                                 bias = 0.f; }

    const float* xrow = x + (size_t)(tok0 + n) * ND;   // A row (M = lane&15)

    v8f acc = {0.f, 0.f, 0.f, 0.f, 0.f, 0.f, 0.f, 0.f};
    #pragma unroll 4
    for (int k0 = 0; k0 < ND; k0 += 4) {
        // 32-bit A 16x4: lane group g holds K = {2g, 2g+1}; B mirrored.
        const v2f a = *(const v2f*)(xrow + k0 + 2 * g);
        const v2f w = *(const v2f*)(wrow + k0 + 2 * g);
        acc = __builtin_amdgcn_wmma_f32_16x16x4_f32(false, a, false, w,
                                                    (short)0, acc, false, false);
    }

    #pragma unroll
    for (int j = 0; j < 8; ++j) {           // D row M = j + 8g -> token tok0+j+8g
        const int tok = tok0 + j + 8 * g;
        const int b = tok / NS, s = tok % NS;
        const float val = acc[j] + bias;
        if (o < NH * NP) {
            const int h = o >> 5, p = o & 31;
            feats[(((size_t)h * NB + b) * NS + s) * NP + p] = (half_t)val;
        } else if (o < NH * NP + NM) {
            logits[(size_t)tok * NM + (o - 128)] = val;
        } else if (o < NOUT) {
            comb[(size_t)tok * NH + (o - 136)] = val;
        }
    }
}

// ---------------------------------------------------------------------------
// Kernel 2: softmax over the M=8 selection logits (one token per thread).
// ---------------------------------------------------------------------------
__global__ void k_softmax(const float* __restrict__ logits, float* __restrict__ probs) {
    const int tok = blockIdx.x * blockDim.x + threadIdx.x;
    if (tok >= NB * NS) return;
    float lg[NM];
    #pragma unroll
    for (int m = 0; m < NM; ++m) lg[m] = logits[(size_t)tok * NM + m];
    float mx = lg[0];
    #pragma unroll
    for (int m = 1; m < NM; ++m) mx = fmaxf(mx, lg[m]);
    float sum = 0.f;
    #pragma unroll
    for (int m = 0; m < NM; ++m) { lg[m] = __expf(lg[m] - mx); sum += lg[m]; }
    const float inv = 1.f / sum;
    #pragma unroll
    for (int m = 0; m < NM; ++m) probs[(size_t)tok * NM + m] = lg[m] * inv;
}

// ---------------------------------------------------------------------------
// Kernel 3: per-token template mix C = sum_m p_m * T_m; t[h] = C @ feat[h] (f16).
// ---------------------------------------------------------------------------
__global__ void k_tmix(const float* __restrict__ probs, const float* __restrict__ templates,
                       const half_t* __restrict__ feats, half_t* __restrict__ tws) {
    __shared__ float Cs[NP * NP];
    __shared__ float pr[NM];
    __shared__ float fv[NH][NP];
    const int tok = blockIdx.x;
    const int b = tok / NS, s = tok % NS;
    const int tid = threadIdx.x;
    if (tid < NM) pr[tid] = probs[(size_t)tok * NM + tid];
    if (tid < NH * NP) {
        const int h = tid >> 5, p = tid & 31;
        fv[h][p] = (float)feats[(((size_t)h * NB + b) * NS + s) * NP + p];
    }
    __syncthreads();
    for (int e = tid; e < NP * NP; e += blockDim.x) {
        float c = 0.f;
        #pragma unroll
        for (int m = 0; m < NM; ++m) c += pr[m] * templates[m * NP * NP + e];
        Cs[e] = c;
    }
    __syncthreads();
    if (tid < NH * NP) {
        const int h = tid >> 5, i = tid & 31;
        float a = 0.f;
        #pragma unroll 8
        for (int j = 0; j < NP; ++j) a += Cs[i * NP + j] * fv[h][j];
        tws[(((size_t)h * NB + b) * NS + s) * NP + i] = (half_t)a;
    }
}

// ---------------------------------------------------------------------------
// Kernel 4 (pass 1): scores computed TRANSPOSED (D row = key s, col = query q)
// so each lane owns one query column -> row stats are pure register reductions.
// ---------------------------------------------------------------------------
__global__ void k_pass1(const half_t* __restrict__ tws, const half_t* __restrict__ feats,
                        float* __restrict__ Mws, float* __restrict__ Lws) {
    __shared__ float sm[4][2][16], sl[4][2][16];
    const int id = blockIdx.x;
    const int qt = id % QT, b = (id / QT) % NB, h = id / (QT * NB);
    const int q0 = qt * 16;
    const int wave = threadIdx.x >> 5;
    const int lane = threadIdx.x & 31;
    const int g = lane >> 4, n = lane & 15;
    const int q = q0 + n;                   // this lane's query column
    const half_t* tbase = tws + (((size_t)h * NB + b) * NS) * NP;
    const half_t* fbase = feats + (((size_t)h * NB + b) * NS) * NP;
    const v16h bfr = load_bfrag(tbase + (size_t)q * NP, g);   // B = t (queries)

    float mr = -__builtin_inff(), lr = 0.f;

    for (int st = wave; st <= qt; st += 4) {   // wave-uniform loop, EXEC stays full
        const int s0 = st * 16;
        const v16h afr = load_afrag(fbase + (size_t)(s0 + n) * NP, g);  // A = feats (keys)
        v8f d = {0.f, 0.f, 0.f, 0.f, 0.f, 0.f, 0.f, 0.f};
        d = __builtin_amdgcn_wmma_f32_16x16x32_f16(false, afr, false, bfr,
                                                   (short)0, d, false, false);
        // element j: score(q, s = s0 + 8g + j)
        float vm = -__builtin_inff();
        #pragma unroll
        for (int j = 0; j < 8; ++j)
            if (s0 + 8 * g + j <= q) vm = fmaxf(vm, d[j]);
        const float mn = fmaxf(mr, vm);
        float e = 0.f;
        #pragma unroll
        for (int j = 0; j < 8; ++j)
            e += (s0 + 8 * g + j <= q) ? __expf(d[j] - mn) : 0.f;
        lr = lr * __expf(fminf(mr - mn, 0.f)) + e;   // fminf kills -inf - -inf NaN
        mr = mn;
    }
    sm[wave][g][n] = mr;
    sl[wave][g][n] = lr;
    __syncthreads();
    if (threadIdx.x < 16) {                 // combine 4 waves x 2 lane-groups
        const int row = threadIdx.x;
        float m = -__builtin_inff();
        for (int w = 0; w < 4; ++w)
            for (int gg = 0; gg < 2; ++gg) m = fmaxf(m, sm[w][gg][row]);
        float l = 0.f;
        for (int w = 0; w < 4; ++w)
            for (int gg = 0; gg < 2; ++gg)
                l += sl[w][gg][row] * __expf(fminf(sm[w][gg][row] - m, 0.f));
        const size_t idx = ((size_t)h * NB + b) * NS + q0 + row;
        Mws[idx] = m;
        Lws[idx] = l;
    }
}

// ---------------------------------------------------------------------------
// Kernel 5 (pass 2): recompute transposed scores (4 heads), combine, write out.
// Lane owns query q -> 8 outputs contiguous in s -> two b128 stores per tile.
// ---------------------------------------------------------------------------
__global__ void k_pass2(const half_t* __restrict__ tws, const half_t* __restrict__ feats,
                        const float* __restrict__ Mws, const float* __restrict__ Lws,
                        const float* __restrict__ comb, float* __restrict__ out) {
    const int id = blockIdx.x;
    const int qt = id % QT, b = id / QT;
    const int q0 = qt * 16;
    const int wave = threadIdx.x >> 5;
    const int lane = threadIdx.x & 31;
    const int g = lane >> 4, n = lane & 15;
    const int q = q0 + n;

    v16h bfr[NH];
    float mh[NH], wgt[NH];
    #pragma unroll
    for (int h = 0; h < NH; ++h) {
        bfr[h] = load_bfrag(tws + (((size_t)h * NB + b) * NS + q) * NP, g);
        const size_t idx = ((size_t)h * NB + b) * NS + q;
        mh[h]  = Mws[idx];
        wgt[h] = comb[(size_t)(b * NS + q) * NH + h] / Lws[idx];
    }

    float* orow = out + ((size_t)b * NS + q) * NS;   // this lane's output row

    for (int st = wave; st < QT; st += 4) {          // wave-uniform
        const int s0 = st * 16;
        if (s0 > q0 + 15) {                          // strictly above diagonal: zeros
            const v4f z = {0.f, 0.f, 0.f, 0.f};
            *(v4f*)(orow + s0 + 8 * g)     = z;
            *(v4f*)(orow + s0 + 8 * g + 4) = z;
            continue;
        }
        v8f d[NH];
        #pragma unroll
        for (int h = 0; h < NH; ++h) {
            const v16h afr =
                load_afrag(feats + (((size_t)h * NB + b) * NS + s0 + n) * NP, g);
            v8f c = {0.f, 0.f, 0.f, 0.f, 0.f, 0.f, 0.f, 0.f};
            d[h] = __builtin_amdgcn_wmma_f32_16x16x32_f16(false, afr, false, bfr[h],
                                                          (short)0, c, false, false);
        }
        Out8 o;
        #pragma unroll
        for (int j = 0; j < 8; ++j) {
            const int s = s0 + 8 * g + j;
            float a = 0.f;
            if (s <= q) {
                #pragma unroll
                for (int h = 0; h < NH; ++h)
                    a += wgt[h] * __expf(d[h][j] - mh[h]);
            }
            o.f[j] = a;
        }
        *(v4f*)(orow + s0 + 8 * g)     = o.v[0];
        *(v4f*)(orow + s0 + 8 * g + 4) = o.v[1];
    }
}

// ---------------------------------------------------------------------------
extern "C" void kernel_launch(void* const* d_in, const int* in_sizes, int n_in,
                              void* d_out, int out_size, void* d_ws, size_t ws_size,
                              hipStream_t stream) {
    (void)in_sizes; (void)n_in; (void)out_size; (void)ws_size;
    const float* x         = (const float*)d_in[0];
    const float* gen_w     = (const float*)d_in[1];
    const float* gen_b     = (const float*)d_in[2];
    const float* sel_w     = (const float*)d_in[3];
    const float* sel_b     = (const float*)d_in[4];
    const float* templates = (const float*)d_in[5];
    const float* comb_w    = (const float*)d_in[6];
    const float* comb_b    = (const float*)d_in[7];
    float* out = (float*)d_out;

    char* ws = (char*)d_ws;
    size_t off = 0;
    half_t* feats  = (half_t*)(ws + off); off += (size_t)NH * NB * NS * NP * sizeof(half_t);
    half_t* tws    = (half_t*)(ws + off); off += (size_t)NH * NB * NS * NP * sizeof(half_t);
    float*  logits = (float*)(ws + off);  off += (size_t)NB * NS * NM * sizeof(float);
    float*  probs  = (float*)(ws + off);  off += (size_t)NB * NS * NM * sizeof(float);
    float*  combv  = (float*)(ws + off);  off += (size_t)NB * NS * NH * sizeof(float);
    float*  Mws    = (float*)(ws + off);  off += (size_t)NH * NB * NS * sizeof(float);
    float*  Lws    = (float*)(ws + off);  off += (size_t)NH * NB * NS * sizeof(float);

    k_feat_wmma<<<(NB * NS / 16) * OTILES, 32, 0, stream>>>(
        x, gen_w, gen_b, sel_w, sel_b, comb_w, comb_b, feats, logits, combv);
    k_softmax<<<(NB * NS + 255) / 256, 256, 0, stream>>>(logits, probs);
    k_tmix<<<NB * NS, 256, 0, stream>>>(probs, templates, feats, tws);
    k_pass1<<<NH * NB * QT, 128, 0, stream>>>(tws, feats, Mws, Lws);
    k_pass2<<<NB * QT, 128, 0, stream>>>(tws, feats, Mws, Lws, combv, out);
}
